// Head_40407052320800
// MI455X (gfx1250) — matrix operand order repaired and checked by reading, not compiled, not run on hardware
//
#include <hip/hip_runtime.h>
#include <hip/hip_bf16.h>

// ---------------------------------------------------------------------------
// Problem constants (match reference)
// ---------------------------------------------------------------------------
static constexpr int B  = 8;
static constexpr int T  = 2048;
static constexpr int C  = 1024;
static constexpr int HS = 64;

typedef __attribute__((ext_vector_type(16))) __bf16 v16bf;
typedef __attribute__((ext_vector_type(8)))  float  v8f;

union FragU  { v16bf v; unsigned int u[8]; uint4 q[2]; };
union Pack2  { __bf16 h[2]; unsigned int u; };
union Pack8  { __bf16 h[8]; uint4 v; };

// ---------------------------------------------------------------------------
// Workspace layout (bytes)
// ---------------------------------------------------------------------------
static constexpr size_t SZ_W   = (size_t)C * HS * 2;        // bf16 weight (transposed [h][c])
static constexpr size_t OFF_WQ = 0;
static constexpr size_t OFF_WK = OFF_WQ + SZ_W;
static constexpr size_t OFF_WV = OFF_WK + SZ_W;
static constexpr size_t SZ_QK  = (size_t)B * T * HS * 2;    // bf16 activations
static constexpr size_t OFF_Q  = OFF_WV + SZ_W;
static constexpr size_t OFF_K  = OFF_Q + SZ_QK;
static constexpr size_t OFF_V  = OFF_K + SZ_QK;             // fp32 v
static constexpr size_t OFF_M  = OFF_V + (size_t)B * T * HS * 4;
static constexpr size_t OFF_S  = OFF_M + (size_t)B * T * 4;

// ---------------------------------------------------------------------------
// WMMA helpers
// ---------------------------------------------------------------------------
__device__ __forceinline__ v8f wmma_bf16(v16bf a, v16bf b, v8f c) {
  // (neg_a, A, neg_b, B, c_mod, C, reuse_a, reuse_b)
  return __builtin_amdgcn_wmma_f32_16x16x32_bf16(false, a, false, b, (short)0, c,
                                                 false, false);
}

// A-fragment (16x32) from LDS tile stored row-major [M][K].
// ISA 7.12.2 packing: lanes 0-15 hold K {0..7,16..23}, lanes 16-31 K {8..15,24..31}.
// ld must be a multiple of 8 so both 16B runs are 16B-aligned -> 2x ds_load_b128.
__device__ __forceinline__ v16bf load_frag_a(const __bf16* p, int ld) {
  int lane = threadIdx.x & 31;
  int m = lane & 15, half = lane >> 4;
  const __bf16* row = p + m * ld + (half << 3);
  FragU f;
  f.q[0] = *(const uint4*)&row[0];     // K base+0..7
  f.q[1] = *(const uint4*)&row[16];    // K base+16..23
  return f.v;
}

// B-fragment (32x16) from LDS tile stored row-major [N][K].
// Same K packing as A: lanes 0-15 hold K {0..7,16..23}, lanes 16-31 K {8..15,24..31}.
// Two 16B runs -> 2x ds_load_b128.
__device__ __forceinline__ v16bf load_frag_b_nk(const __bf16* p, int ld) {
  int lane = threadIdx.x & 31;
  int n = lane & 15, half = lane >> 4;
  const __bf16* row = p + n * ld + (half << 3);
  FragU f;
  f.q[0] = *(const uint4*)&row[0];
  f.q[1] = *(const uint4*)&row[16];
  return f.v;
}

__device__ __forceinline__ v8f zero8() { v8f z = {0,0,0,0,0,0,0,0}; return z; }

// ---------------------------------------------------------------------------
// Kernel 0: fp32 -> bf16 weights, TRANSPOSED to [h][c]; fold 1/sqrt(C)=1/32
// (exact power of two) into Wq.
// ---------------------------------------------------------------------------
__global__ void wconv_kernel(const float* Wk, const float* Wq, const float* Wv,
                             __bf16* wkt, __bf16* wqt, __bf16* wvt) {
  int i = blockIdx.x * blockDim.x + threadIdx.x;
  if (i < C * HS) {
    int c = i / HS, h = i % HS;          // reads coalesced over h
    size_t o = (size_t)h * C + c;
    wkt[o] = (__bf16)Wk[i];
    wqt[o] = (__bf16)(Wq[i] * 0.03125f);
    wvt[o] = (__bf16)Wv[i];
  }
}

// ---------------------------------------------------------------------------
// Kernel 1: projections, computed transposed: D[h][t] = Wt[h][c] * x^T[c][t].
// 64 x-rows per block, 8 waves, 48 16x16 tiles (3 mats x 4 hc x 4 tr).
// ---------------------------------------------------------------------------
__global__ __launch_bounds__(256) void proj_kernel(
    const float* __restrict__ x, const __bf16* __restrict__ wqt,
    const __bf16* __restrict__ wkt, const __bf16* __restrict__ wvt,
    __bf16* __restrict__ q, __bf16* __restrict__ k, float* __restrict__ v) {
  __shared__ __bf16 xs[64][40];        // [t][c] chunk, ld 40 (pad 8)
  __shared__ __bf16 wsm[3][64][40];    // [mat][h][c] chunk

  int tid = threadIdx.x, wave = tid >> 5, lane = tid & 31;
  int row0 = blockIdx.x * 64;

  v8f acc[6];
#pragma unroll
  for (int i = 0; i < 6; ++i) acc[i] = zero8();

  for (int kc = 0; kc < C; kc += 32) {
    // stage x chunk: 64 rows x 32 cols, fp32 -> bf16, 8B stores
    for (int i = tid; i < 64 * 8; i += 256) {
      int r = i >> 3, c4 = i & 7;
      float4 f = *(const float4*)&x[(size_t)(row0 + r) * C + kc + c4 * 4];
      Pack2 p0; p0.h[0] = (__bf16)f.x; p0.h[1] = (__bf16)f.y;
      Pack2 p1; p1.h[0] = (__bf16)f.z; p1.h[1] = (__bf16)f.w;
      unsigned int* dst = (unsigned int*)&xs[r][c4 * 4];
      dst[0] = p0.u; dst[1] = p1.u;
    }
    // stage Wt chunks: 3 x 64 rows x 32 cols bf16, uint4 copies
    for (int i = tid; i < 3 * 64 * 4; i += 256) {
      int mat = i >> 8;
      int rem = i & 255;
      int n = rem >> 2, c4 = rem & 3;
      const __bf16* wsrc = (mat == 0) ? wqt : (mat == 1) ? wkt : wvt;
      *(uint4*)&wsm[mat][n][c4 * 8] =
          *(const uint4*)&wsrc[(size_t)n * C + kc + c4 * 8];
    }
    __syncthreads();
#pragma unroll
    for (int i = 0; i < 6; ++i) {
      int tile = wave + 8 * i;            // 0..47
      int mat = tile >> 4, rem = tile & 15;
      int hc = rem >> 2, tr = rem & 3;
      v16bf a = load_frag_a(&wsm[mat][hc * 16][0], 40);    // Wt[h][c]
      v16bf b = load_frag_b_nk(&xs[tr * 16][0], 40);       // x^T as [N=t][K=c]
      acc[i] = wmma_bf16(a, b, acc[i]);
    }
    __syncthreads();
  }

  int tl = lane & 15, half = lane >> 4;
#pragma unroll
  for (int i = 0; i < 6; ++i) {
    int tile = wave + 8 * i;
    int mat = tile >> 4, rem = tile & 15;
    int hc = rem >> 2, tr = rem & 3;
    int trow = row0 + tr * 16 + tl;
    int hbase = hc * 16 + 8 * half;
    if (mat == 2) {                       // v: fp32, two float4
      size_t o = (size_t)trow * HS + hbase;
      float4 lo = {acc[i][0], acc[i][1], acc[i][2], acc[i][3]};
      float4 hi = {acc[i][4], acc[i][5], acc[i][6], acc[i][7]};
      *(float4*)&v[o]     = lo;
      *(float4*)&v[o + 4] = hi;
    } else {                              // q/k: bf16, one uint4
      Pack8 p;
#pragma unroll
      for (int r = 0; r < 8; ++r) p.h[r] = (__bf16)acc[i][r];
      __bf16* dst = (mat == 0) ? q : k;
      *(uint4*)&dst[(size_t)trow * HS + hbase] = p.v;
    }
  }
}

// ---------------------------------------------------------------------------
// Kernel 2: per-key-column softmax stats (softmax over the QUERY axis, per
// reference axis=1). Block owns 64 columns; two sweeps over query rows:
// pass 1 column max M[s], pass 2 column sum S[s] of exp(.-M).
// ---------------------------------------------------------------------------
__global__ __launch_bounds__(256) void stats_kernel(
    const __bf16* __restrict__ q, const __bf16* __restrict__ kmat,
    float* __restrict__ Mout, float* __restrict__ Sout) {
  __shared__ __bf16 Ks[64][72];   // [s][h]
  __shared__ __bf16 Qs[128][72];  // [t][h]
  __shared__ float  red[8][64];
  __shared__ float  colM[64];

  int tid = threadIdx.x, wave = tid >> 5, lane = tid & 31;
  int b  = blockIdx.y;
  int s0 = blockIdx.x * 64;
  const __bf16* qb = q    + (size_t)b * T * HS;
  const __bf16* kb = kmat + (size_t)b * T * HS;

  for (int i = tid; i < 64 * 8; i += 256) {
    int r = i >> 3, c4 = i & 7;
    *(uint4*)&Ks[r][c4 * 8] = *(const uint4*)&kb[(size_t)(s0 + r) * HS + c4 * 8];
  }
  __syncthreads();

  v16bf kb0[4], kb1[4];
#pragma unroll
  for (int nc = 0; nc < 4; ++nc) {
    kb0[nc] = load_frag_b_nk(&Ks[nc * 16][0], 72);
    kb1[nc] = load_frag_b_nk(&Ks[nc * 16][32], 72);
  }

  int tbase = s0 & ~127;
  int tw = wave * 16;
  int half = lane >> 4;

  // ---- pass 1: column max ----
  float maxv[4] = {-__builtin_inff(), -__builtin_inff(),
                   -__builtin_inff(), -__builtin_inff()};
  for (int t0 = tbase; t0 < T; t0 += 128) {
    __syncthreads();
    for (int i = tid; i < 128 * 8; i += 256) {
      int r = i >> 3, c4 = i & 7;
      *(uint4*)&Qs[r][c4 * 8] = *(const uint4*)&qb[(size_t)(t0 + r) * HS + c4 * 8];
    }
    if (t0 + 128 < T)
      __builtin_prefetch(&qb[(size_t)(t0 + 128) * HS + (tid & 63) * 32], 0, 0);
    __syncthreads();
    v16bf a0 = load_frag_a(&Qs[tw][0], 72);
    v16bf a1 = load_frag_a(&Qs[tw][32], 72);
#pragma unroll
    for (int nc = 0; nc < 4; ++nc) {
      v8f c = zero8();
      c = wmma_bf16(a0, kb0[nc], c);
      c = wmma_bf16(a1, kb1[nc], c);
      int sg = s0 + nc * 16 + (lane & 15);
#pragma unroll
      for (int r = 0; r < 8; ++r) {
        int tg = t0 + tw + r + 8 * half;
        if (tg >= sg) maxv[nc] = fmaxf(maxv[nc], c[r]);
      }
    }
  }
#pragma unroll
  for (int nc = 0; nc < 4; ++nc)
    maxv[nc] = fmaxf(maxv[nc], __shfl_xor(maxv[nc], 16, 32));
  if (lane < 16)
#pragma unroll
    for (int nc = 0; nc < 4; ++nc) red[wave][nc * 16 + lane] = maxv[nc];
  __syncthreads();
  if (tid < 64) {
    float m = red[0][tid];
#pragma unroll
    for (int w2 = 1; w2 < 8; ++w2) m = fmaxf(m, red[w2][tid]);
    colM[tid] = m;
    Mout[(size_t)b * T + s0 + tid] = m;
  }
  __syncthreads();

  // ---- pass 2: column sum of exp ----
  float colMv[4];
#pragma unroll
  for (int nc = 0; nc < 4; ++nc) colMv[nc] = colM[nc * 16 + (lane & 15)];
  float sumv[4] = {0.f, 0.f, 0.f, 0.f};
  for (int t0 = tbase; t0 < T; t0 += 128) {
    __syncthreads();
    for (int i = tid; i < 128 * 8; i += 256) {
      int r = i >> 3, c4 = i & 7;
      *(uint4*)&Qs[r][c4 * 8] = *(const uint4*)&qb[(size_t)(t0 + r) * HS + c4 * 8];
    }
    __syncthreads();
    v16bf a0 = load_frag_a(&Qs[tw][0], 72);
    v16bf a1 = load_frag_a(&Qs[tw][32], 72);
#pragma unroll
    for (int nc = 0; nc < 4; ++nc) {
      v8f c = zero8();
      c = wmma_bf16(a0, kb0[nc], c);
      c = wmma_bf16(a1, kb1[nc], c);
      int sg = s0 + nc * 16 + (lane & 15);
#pragma unroll
      for (int r = 0; r < 8; ++r) {
        int tg = t0 + tw + r + 8 * half;
        if (tg >= sg) sumv[nc] += __expf(c[r] - colMv[nc]);
      }
    }
  }
#pragma unroll
  for (int nc = 0; nc < 4; ++nc) sumv[nc] += __shfl_xor(sumv[nc], 16, 32);
  if (lane < 16)
#pragma unroll
    for (int nc = 0; nc < 4; ++nc) red[wave][nc * 16 + lane] = sumv[nc];
  __syncthreads();
  if (tid < 64) {
    float s = 0.f;
#pragma unroll
    for (int w2 = 0; w2 < 8; ++w2) s += red[w2][tid];
    Sout[(size_t)b * T + s0 + tid] = s;
  }
}

// ---------------------------------------------------------------------------
// Kernel 3: out[t,:] = sum_{s<=t} exp(score-M[s]) * (v[s,:]/S[s]).
//   score^T = K*Q^T            (D tile = [s][t])
//   P stored as Ps[t][s]       (single ds_store_b128 per tile per thread)
//   out^T   = V^T * P^T        (D tile = [h][t], contiguous float4 out stores)
// ---------------------------------------------------------------------------
__global__ __launch_bounds__(256) void out_kernel(
    const __bf16* __restrict__ q, const __bf16* __restrict__ kmat,
    const float* __restrict__ v, const float* __restrict__ M,
    const float* __restrict__ S, float* __restrict__ out) {
  __shared__ __bf16 Qs[128][72];     // [t][h]
  __shared__ __bf16 Ks[64][72];      // [s][h]
  __shared__ __bf16 Vst[64][72];     // [h][s], pre-scaled by 1/S[s]
  __shared__ float  Ms[64];
  __shared__ __bf16 Ps[8][16][72];   // per-wave, [t][s]

  int tid = threadIdx.x, wave = tid >> 5, lane = tid & 31;
  int b  = blockIdx.y;
  int t0 = blockIdx.x * 128;
  const __bf16* qb = q    + (size_t)b * T * HS;
  const __bf16* kb = kmat + (size_t)b * T * HS;
  const float*  vb = v + (size_t)b * T * HS;
  const float*  Mb = M + (size_t)b * T;
  const float*  Sb = S + (size_t)b * T;

  for (int i = tid; i < 128 * 8; i += 256) {
    int r = i >> 3, c4 = i & 7;
    *(uint4*)&Qs[r][c4 * 8] = *(const uint4*)&qb[(size_t)(t0 + r) * HS + c4 * 8];
  }
  __syncthreads();

  int tw = wave * 16;
  int tl = lane & 15, half = lane >> 4;
  int tg = t0 + tw + tl;

  // Q fragments are loop-invariant (Qs staged once)
  v16bf qf0 = load_frag_b_nk(&Qs[tw][0], 72);
  v16bf qf1 = load_frag_b_nk(&Qs[tw][32], 72);

  v8f acc[4];
#pragma unroll
  for (int i = 0; i < 4; ++i) acc[i] = zero8();

  int smax = t0 + 127;
  for (int s0 = 0; s0 <= smax; s0 += 64) {
    __syncthreads();  // prior reads of Ks/Vst finished
    for (int i = tid; i < 64 * 8; i += 256) {
      int r = i >> 3, c4 = i & 7;
      *(uint4*)&Ks[r][c4 * 8] = *(const uint4*)&kb[(size_t)(s0 + r) * HS + c4 * 8];
    }
    // stage V transposed + scaled: 512 groups of (2 s) x (4 h); coalesced
    // float4 reads, packed b32 LDS stores.
    for (int i = tid; i < 512; i += 256) {
      int sp = i >> 4, hp = i & 15;
      int s = sp * 2, h = hp * 4;
      float inv0 = 1.0f / Sb[s0 + s];
      float inv1 = 1.0f / Sb[s0 + s + 1];
      float4 f0 = *(const float4*)&vb[(size_t)(s0 + s) * HS + h];
      float4 f1 = *(const float4*)&vb[(size_t)(s0 + s + 1) * HS + h];
      float a0[4] = {f0.x, f0.y, f0.z, f0.w};
      float a1[4] = {f1.x, f1.y, f1.z, f1.w};
#pragma unroll
      for (int j = 0; j < 4; ++j) {
        Pack2 pk;
        pk.h[0] = (__bf16)(a0[j] * inv0);
        pk.h[1] = (__bf16)(a1[j] * inv1);
        *(unsigned int*)&Vst[h + j][s] = pk.u;
      }
    }
    if (tid < 64) Ms[tid] = Mb[s0 + tid];
    if (s0 + 64 <= smax) {
      __builtin_prefetch(&kb[(size_t)(s0 + 64) * HS + (tid & 63) * 32], 0, 0);
      __builtin_prefetch(&vb[(size_t)(s0 + 64) * HS + (tid & 63) * 64], 0, 0);
    }
    __syncthreads();

    // score^T tiles: D[s][t]
#pragma unroll
    for (int nc = 0; nc < 4; ++nc) {
      v16bf ka0 = load_frag_a(&Ks[nc * 16][0], 72);
      v16bf ka1 = load_frag_a(&Ks[nc * 16][32], 72);
      v8f c = zero8();
      c = wmma_bf16(ka0, qf0, c);
      c = wmma_bf16(ka1, qf1, c);
      // thread holds s rows: s_loc = nc*16 + 8*half + r; column t = tg
      int sbase = nc * 16 + 8 * half;
      Pack8 pk;
#pragma unroll
      for (int r = 0; r < 8; ++r) {
        int sl = sbase + r;
        float p = (tg >= s0 + sl) ? __expf(c[r] - Ms[sl]) : 0.0f;
        pk.h[r] = (__bf16)p;
      }
      *(uint4*)&Ps[wave][tl][sbase] = pk.v;   // single ds_store_b128
    }

    // out^T accumulation: D[h][t] += Vst[h][s] * Ps^T[s][t]
    // (Ps[wave] written and read by the same wave; DS ops are in-order)
    v16bf pb0 = load_frag_b_nk(&Ps[wave][0][0], 72);
    v16bf pb1 = load_frag_b_nk(&Ps[wave][0][32], 72);
#pragma unroll
    for (int hc = 0; hc < 4; ++hc) {
      v16bf va0 = load_frag_a(&Vst[hc * 16][0], 72);
      v16bf va1 = load_frag_a(&Vst[hc * 16][32], 72);
      acc[hc] = wmma_bf16(va0, pb0, acc[hc]);
      acc[hc] = wmma_bf16(va1, pb1, acc[hc]);
    }
  }

  // store: thread owns query row tg, 8 contiguous h per tile
#pragma unroll
  for (int hc = 0; hc < 4; ++hc) {
    int hbase = hc * 16 + 8 * half;
    size_t o = ((size_t)b * T + tg) * HS + hbase;
    float4 lo = {acc[hc][0], acc[hc][1], acc[hc][2], acc[hc][3]};
    float4 hi = {acc[hc][4], acc[hc][5], acc[hc][6], acc[hc][7]};
    *(float4*)&out[o]     = lo;
    *(float4*)&out[o + 4] = hi;
  }
}

// ---------------------------------------------------------------------------
// Launch
// ---------------------------------------------------------------------------
extern "C" void kernel_launch(void* const* d_in, const int* in_sizes, int n_in,
                              void* d_out, int out_size, void* d_ws, size_t ws_size,
                              hipStream_t stream) {
  const float* x  = (const float*)d_in[0];
  const float* Wk = (const float*)d_in[1];
  const float* Wq = (const float*)d_in[2];
  const float* Wv = (const float*)d_in[3];

  char* ws = (char*)d_ws;
  __bf16* wqt = (__bf16*)(ws + OFF_WQ);
  __bf16* wkt = (__bf16*)(ws + OFF_WK);
  __bf16* wvt = (__bf16*)(ws + OFF_WV);
  __bf16* qb  = (__bf16*)(ws + OFF_Q);
  __bf16* kbf = (__bf16*)(ws + OFF_K);
  float*  vbf = (float*)(ws + OFF_V);
  float*  Mbf = (float*)(ws + OFF_M);
  float*  Sbf = (float*)(ws + OFF_S);
  float*  out = (float*)d_out;

  wconv_kernel<<<(C * HS + 255) / 256, 256, 0, stream>>>(Wk, Wq, Wv, wkt, wqt, wvt);
  proj_kernel<<<B * T / 64, 256, 0, stream>>>(x, wqt, wkt, wvt, qb, kbf, vbf);
  stats_kernel<<<dim3(T / 64, B), 256, 0, stream>>>(qb, kbf, Mbf, Sbf);
  out_kernel<<<dim3(T / 128, B), 256, 0, stream>>>(qb, kbf, vbf, Mbf, Sbf, out);
}
